// Chamfer_Loss_82403242541369
// MI455X (gfx1250) — matrix operand order repaired and checked by reading, not compiled, not run on hardware
//
#include <hip/hip_runtime.h>
#include <math.h>

typedef __attribute__((ext_vector_type(16))) _Float16 v16h;
typedef __attribute__((ext_vector_type(8)))  float    v8f;

#define MT 4   // M-tiles (16 rows each) per wave in the chamfer kernel

__device__ __forceinline__ void split_hl(float x, _Float16& h, _Float16& l) {
  h = (_Float16)x;
  l = (_Float16)(x - (float)h);
}

// ---------------------------------------------------------------- utilities
__global__ void zero_kernel(float* __restrict__ p, int n) {
  int i = blockIdx.x * blockDim.x + threadIdx.x;
  if (i < n) p[i] = 0.0f;
}

// dst[b,v,c] = src[b,v+1,c] - src[b,v,c]
__global__ void diff_kernel(const float* __restrict__ src, float* __restrict__ dst,
                            int B, int V) {
  int i = blockIdx.x * blockDim.x + threadIdx.x;
  int n = B * (V - 1) * 3;
  if (i >= n) return;
  int c = i % 3;
  int r = i / 3;
  int v = r % (V - 1);
  int b = r / (V - 1);
  dst[i] = src[((size_t)b * V + v + 1) * 3 + c] - src[((size_t)b * V + v) * 3 + c];
}

// --------------------------------------------- pack Y-side WMMA B-fragments
// B fragment (32x16 f16), lanes 0-15 hold K0-15 for column n = tile*16+lane:
//   K0-2 = hy, K3-5 = ly, K6-8 = hy, K9 = hi(y^2), K10 = lo(y^2)
// Combined with the A packing below, the WMMA emits  2*x.y - y^2  directly.
__global__ void pack_y_kernel(const float* __restrict__ Y, v16h* __restrict__ out,
                              int B, int NY) {
  const int lane = threadIdx.x & 31;
  const int wid  = threadIdx.x >> 5;
  const int tilesN = (NY + 15) >> 4;
  int w = blockIdx.x * (blockDim.x >> 5) + wid;
  if (w >= B * tilesN) return;
  int b  = w / tilesN;
  int nt = w % tilesN;

  v16h bf = {};
  if (lane < 16) {
    int ncol = (nt << 4) + lane;
    if (ncol < NY) {
      const float* q = Y + ((size_t)b * NY + ncol) * 3;
      float y0 = q[0], y1 = q[1], y2 = q[2];
      float y2s = y0 * y0 + y1 * y1 + y2 * y2;
      _Float16 g0, m0, g1, m1, g2, m2;
      split_hl(y0, g0, m0); split_hl(y1, g1, m1); split_hl(y2, g2, m2);
      _Float16 yh, yl;
      split_hl(y2s, yh, yl);
      bf[0] = g0; bf[1] = g1; bf[2] = g2;
      bf[3] = m0; bf[4] = m1; bf[5] = m2;
      bf[6] = g0; bf[7] = g1; bf[8] = g2;
      bf[9] = yh; bf[10] = yl;
    } else {
      bf[9] = (_Float16)65504.0f;   // padded col: D = -65504, never wins max
    }
  }
  out[(size_t)w * 32 + lane] = bf;
}

// ------------------------------------------------- chamfer row-min via WMMA
// WMMA computes D[m][n] = 2*x_m.y_n - |y_n|^2 ; row min of d2 = x^2 - max_n D.
// Accumulates sum over (b, rows of X) of min_n ||x-y||^2 into *acc.
__global__ void chamfer_rowmin_kernel(const float* __restrict__ X,
                                      const v16h* __restrict__ Yp,
                                      int B, int NX, int NY,
                                      float* __restrict__ acc) {
  const int lane = threadIdx.x & 31;
  const int wid  = threadIdx.x >> 5;
  const int tilesM = (NX + 15) >> 4;
  const int tilesN = (NY + 15) >> 4;
  const int groups = (tilesM + MT - 1) / MT;

  int w = blockIdx.x * (blockDim.x >> 5) + wid;
  if (w >= B * groups) return;
  int b = w / groups;
  int g = w % groups;

  // ---- Build MT A fragments (16x32 f16 each).
  // Lanes 0-15 hold K0-15: 2hx0,2hx1,2hx2, 2hx0,2hx1,2hx2, 2lx0,2lx1, K8 = 2lx2, K9 = -1, K10 = -1
  // Lanes 16-31 hold K16-31: zero
  v16h afrag[MT];
  float rmax[MT][8];
#pragma unroll
  for (int j = 0; j < MT; ++j) {
    int mrow = (g * MT + j) * 16 + (lane & 15);
    float x0 = 0.f, x1 = 0.f, x2 = 0.f;
    if (mrow < NX) {
      const float* p = X + ((size_t)b * NX + mrow) * 3;
      x0 = p[0]; x1 = p[1]; x2 = p[2];
    }
    _Float16 h0, l0, h1, l1, h2, l2;
    split_hl(x0, h0, l0); split_hl(x1, h1, l1); split_hl(x2, h2, l2);
    v16h a = {};
    if (lane < 16) {
      _Float16 t0 = h0 + h0, t1 = h1 + h1, t2 = h2 + h2;   // exact doubling
      a[0] = t0; a[1] = t1; a[2] = t2;
      a[3] = t0; a[4] = t1; a[5] = t2;
      a[6] = l0 + l0; a[7] = l1 + l1;
      a[8] = l2 + l2;
      a[9] = (_Float16)(-1.0f);
      a[10] = (_Float16)(-1.0f);
    }
    afrag[j] = a;
#pragma unroll
    for (int r = 0; r < 8; ++r) rmax[j][r] = -3.0e38f;
  }

  const v16h* Ybp = Yp + (size_t)b * tilesN * 32;

  for (int nt = 0; nt < tilesN; ++nt) {
    v16h bfrag = Ybp[(size_t)nt * 32 + lane];
    if (nt + 1 < tilesN)
      __builtin_prefetch(&Ybp[(size_t)(nt + 1) * 32 + lane], 0, 3);
#pragma unroll
    for (int j = 0; j < MT; ++j) {
      v8f c = {};
      c = __builtin_amdgcn_wmma_f32_16x16x32_f16(false, afrag[j], false, bfrag,
                                                 (short)0, c, false, false);
#pragma unroll
      for (int r = 0; r < 8; ++r)
        rmax[j][r] = fmaxf(rmax[j][r], c[r]);
    }
  }

  // max across the 16 columns held by each 16-lane half
#pragma unroll
  for (int j = 0; j < MT; ++j) {
#pragma unroll
    for (int r = 0; r < 8; ++r) {
      float v = rmax[j][r];
      v = fmaxf(v, __shfl_xor(v, 1, 16));
      v = fmaxf(v, __shfl_xor(v, 2, 16));
      v = fmaxf(v, __shfl_xor(v, 4, 16));
      v = fmaxf(v, __shfl_xor(v, 8, 16));
      rmax[j][r] = v;
    }
  }

  // lane 0 -> rows +0..7 ; lane 16 -> rows +8..15 of each M-tile
  if ((lane & 15) == 0) {
    int roff = (lane >> 4) * 8;
    float s = 0.f;
#pragma unroll
    for (int j = 0; j < MT; ++j) {
#pragma unroll
      for (int r = 0; r < 8; ++r) {
        int row = (g * MT + j) * 16 + roff + r;
        if (row < NX) {
          const float* p = X + ((size_t)b * NX + row) * 3;
          float a0 = p[0], a1 = p[1], a2 = p[2];
          s += (a0 * a0 + a1 * a1 + a2 * a2) - rmax[j][r];
        }
      }
    }
    atomicAdd(acc, s);
  }
}

// ---------------------------------------------------------------- edge loss
__global__ void edge_kernel(const float* __restrict__ pred, const int* __restrict__ edges,
                            int B, int V, int E, float* __restrict__ acc) {
  int i = blockIdx.x * blockDim.x + threadIdx.x;
  float val = 0.f;
  if (i < B * E) {
    int b = i / E, e = i % E;
    int e0 = edges[e * 2 + 0], e1 = edges[e * 2 + 1];
    const float* p0 = pred + ((size_t)b * V + e0) * 3;
    const float* p1 = pred + ((size_t)b * V + e1) * 3;
    float d0 = p0[0] - p1[0], d1 = p0[1] - p1[1], d2 = p0[2] - p1[2];
    val = d0 * d0 + d1 * d1 + d2 * d2;
  }
  val += __shfl_xor(val, 1);  val += __shfl_xor(val, 2);
  val += __shfl_xor(val, 4);  val += __shfl_xor(val, 8);
  val += __shfl_xor(val, 16);
  if ((threadIdx.x & 31) == 0) atomicAdd(acc, val);
}

// ------------------------------------------------------------- normal loss
__global__ void normal_kernel(const float* __restrict__ pred, const int* __restrict__ pairs,
                              int B, int V, int P, float* __restrict__ acc) {
  int i = blockIdx.x * blockDim.x + threadIdx.x;
  float val = 0.f;
  if (i < B * P) {
    int b = i / P, p = i % P;
    int i0 = pairs[p * 4 + 0], i1 = pairs[p * 4 + 1];
    int ia = pairs[p * 4 + 2], ib = pairs[p * 4 + 3];
    const float* base = pred + (size_t)b * V * 3;
    float p0x = base[i0 * 3], p0y = base[i0 * 3 + 1], p0z = base[i0 * 3 + 2];
    float p1x = base[i1 * 3], p1y = base[i1 * 3 + 1], p1z = base[i1 * 3 + 2];
    float ax  = base[ia * 3], ay  = base[ia * 3 + 1], az  = base[ia * 3 + 2];
    float bx  = base[ib * 3], by  = base[ib * 3 + 1], bz  = base[ib * 3 + 2];
    float dx = p1x - p0x, dy = p1y - p0y, dz = p1z - p0z;
    float ex = ax - p0x, ey = ay - p0y, ez = az - p0z;
    float fx = bx - p0x, fy = by - p0y, fz = bz - p0z;
    float n0x = dy * ez - dz * ey, n0y = dz * ex - dx * ez, n0z = dx * ey - dy * ex;
    float n1x = -(dy * fz - dz * fy), n1y = -(dz * fx - dx * fz), n1z = -(dx * fy - dy * fx);
    float dot = n0x * n1x + n0y * n1y + n0z * n1z;
    float m0 = sqrtf(n0x * n0x + n0y * n0y + n0z * n0z);
    float m1 = sqrtf(n1x * n1x + n1y * n1y + n1z * n1z);
    const float eps = 1e-8f;
    float cosv = dot / (fmaxf(m0, eps) * fmaxf(m1, eps));
    val = 1.0f - cosv;
  }
  val += __shfl_xor(val, 1);  val += __shfl_xor(val, 2);
  val += __shfl_xor(val, 4);  val += __shfl_xor(val, 8);
  val += __shfl_xor(val, 16);
  if ((threadIdx.x & 31) == 0) atomicAdd(acc, val);
}

// ---------------------------------------------- laplacian: cotangent scatter
__global__ void lap_face_kernel(const float* __restrict__ pred, const int* __restrict__ faces,
                                int B, int V, int F,
                                float* __restrict__ Lx, float* __restrict__ rs) {
  int i = blockIdx.x * blockDim.x + threadIdx.x;
  if (i >= B * F) return;
  int b = i / F, f = i % F;
  int i0 = faces[f * 3 + 0], i1 = faces[f * 3 + 1], i2 = faces[f * 3 + 2];
  const float* vb = pred + (size_t)b * V * 3;
  float v0x = vb[i0 * 3], v0y = vb[i0 * 3 + 1], v0z = vb[i0 * 3 + 2];
  float v1x = vb[i1 * 3], v1y = vb[i1 * 3 + 1], v1z = vb[i1 * 3 + 2];
  float v2x = vb[i2 * 3], v2y = vb[i2 * 3 + 1], v2z = vb[i2 * 3 + 2];
  float Ax = v1x - v2x, Ay = v1y - v2y, Az = v1z - v2z;
  float Bx = v0x - v2x, By = v0y - v2y, Bz = v0z - v2z;
  float Cx = v0x - v1x, Cy = v0y - v1y, Cz = v0z - v1z;
  float A = sqrtf(Ax * Ax + Ay * Ay + Az * Az);
  float Bl = sqrtf(Bx * Bx + By * By + Bz * Bz);
  float C = sqrtf(Cx * Cx + Cy * Cy + Cz * Cz);
  float s = 0.5f * (A + Bl + C);
  float area = sqrtf(fmaxf(s * (s - A) * (s - Bl) * (s - C), 1e-12f));
  float A2 = A * A, B2 = Bl * Bl, C2 = C * C;
  float w0 = 0.25f * (B2 + C2 - A2) / area;
  float w1 = 0.25f * (A2 + C2 - B2) / area;
  float w2 = 0.25f * (A2 + B2 - C2) / area;

  float* Lb = Lx + (size_t)b * V * 3;
  float* rb = rs + (size_t)b * V;
  atomicAdd(&Lb[i1 * 3 + 0], w0 * v2x); atomicAdd(&Lb[i1 * 3 + 1], w0 * v2y); atomicAdd(&Lb[i1 * 3 + 2], w0 * v2z);
  atomicAdd(&Lb[i2 * 3 + 0], w0 * v1x); atomicAdd(&Lb[i2 * 3 + 1], w0 * v1y); atomicAdd(&Lb[i2 * 3 + 2], w0 * v1z);
  atomicAdd(&rb[i1], w0); atomicAdd(&rb[i2], w0);

  atomicAdd(&Lb[i2 * 3 + 0], w1 * v0x); atomicAdd(&Lb[i2 * 3 + 1], w1 * v0y); atomicAdd(&Lb[i2 * 3 + 2], w1 * v0z);
  atomicAdd(&Lb[i0 * 3 + 0], w1 * v2x); atomicAdd(&Lb[i0 * 3 + 1], w1 * v2y); atomicAdd(&Lb[i0 * 3 + 2], w1 * v2z);
  atomicAdd(&rb[i2], w1); atomicAdd(&rb[i0], w1);

  atomicAdd(&Lb[i0 * 3 + 0], w2 * v1x); atomicAdd(&Lb[i0 * 3 + 1], w2 * v1y); atomicAdd(&Lb[i0 * 3 + 2], w2 * v1z);
  atomicAdd(&Lb[i1 * 3 + 0], w2 * v0x); atomicAdd(&Lb[i1 * 3 + 1], w2 * v0y); atomicAdd(&Lb[i1 * 3 + 2], w2 * v0z);
  atomicAdd(&rb[i0], w2); atomicAdd(&rb[i1], w2);
}

__global__ void lap_final_kernel(const float* __restrict__ pred,
                                 const float* __restrict__ Lx, const float* __restrict__ rs,
                                 int B, int V, float* __restrict__ acc) {
  int i = blockIdx.x * blockDim.x + threadIdx.x;
  float val = 0.f;
  if (i < B * V) {
    float r = rs[i];
    float nw = (r > 0.f) ? (1.0f / r) : 0.0f;
    float ex = Lx[(size_t)i * 3 + 0] * nw - pred[(size_t)i * 3 + 0];
    float ey = Lx[(size_t)i * 3 + 1] * nw - pred[(size_t)i * 3 + 1];
    float ez = Lx[(size_t)i * 3 + 2] * nw - pred[(size_t)i * 3 + 2];
    val = sqrtf(ex * ex + ey * ey + ez * ez);
  }
  val += __shfl_xor(val, 1);  val += __shfl_xor(val, 2);
  val += __shfl_xor(val, 4);  val += __shfl_xor(val, 8);
  val += __shfl_xor(val, 16);
  if ((threadIdx.x & 31) == 0) atomicAdd(acc, val);
}

// ---------------------------------------------------------------- combine
__global__ void final_kernel(const float* __restrict__ acc, float* __restrict__ out,
                             int B, int V, int E, int P) {
  float cham = acc[0] / ((float)B * (float)V);
  float vel  = acc[1] / ((float)B * (float)(V - 1));
  float edge = acc[2] / ((float)B * (float)E);
  float nrm  = acc[3] / ((float)B * (float)P);
  float lap  = acc[4] / ((float)B * (float)V);
  out[0] = cham + 0.05f * lap + 0.01f * nrm + 0.5f * edge + 10.0f * vel;
}

// ---------------------------------------------------------------- launcher
extern "C" void kernel_launch(void* const* d_in, const int* in_sizes, int n_in,
                              void* d_out, int out_size, void* d_ws, size_t ws_size,
                              hipStream_t stream) {
  const float* pred  = (const float*)d_in[0];
  const float* targ  = (const float*)d_in[1];
  const int*   faces = (const int*)d_in[2];
  const int*   edges = (const int*)d_in[3];
  const int*   pairs = (const int*)d_in[4];

  const int V = 71 * 71;
  const int B = in_sizes[0] / (V * 3);
  const int F = in_sizes[2] / 3;
  const int E = in_sizes[3] / 2;
  const int P = in_sizes[4] / 4;
  const int Nv = V - 1;

  const int tilesP = (V + 15) / 16;    // 316
  const int tilesV = (Nv + 15) / 16;   // 315

  // ---- workspace layout (packed fragments first, 32B-aligned) ----
  char* base = (char*)d_ws;
  size_t packElemsP = (size_t)B * tilesP * 32;   // v16h units (32B each)
  size_t packElemsV = (size_t)B * tilesV * 32;
  v16h* packPred  = (v16h*)base;                       base += packElemsP * 32;
  v16h* packTarg  = (v16h*)base;                       base += packElemsP * 32;
  v16h* packDPred = (v16h*)base;                       base += packElemsV * 32;
  v16h* packDTarg = (v16h*)base;                       base += packElemsV * 32;
  float* acc   = (float*)base;                 // 8 floats
  float* Lx    = acc + 8;                      // B*V*3
  float* rs    = Lx + (size_t)B * V * 3;       // B*V
  float* dpred = rs + (size_t)B * V;           // B*Nv*3
  float* dtarg = dpred + (size_t)B * Nv * 3;

  // zero accumulators + laplacian scratch every call
  int nzero = 8 + B * V * 3 + B * V;
  zero_kernel<<<(nzero + 255) / 256, 256, 0, stream>>>(acc, nzero);

  // velocity fields
  int nd = B * Nv * 3;
  diff_kernel<<<(nd + 255) / 256, 256, 0, stream>>>(pred, dpred, B, V);
  diff_kernel<<<(nd + 255) / 256, 256, 0, stream>>>(targ, dtarg, B, V);

  // pack Y-side fragments (4 waves / block)
  {
    int wavesP = B * tilesP, wavesV = B * tilesV;
    pack_y_kernel<<<(wavesP + 3) / 4, 128, 0, stream>>>(pred,  packPred,  B, V);
    pack_y_kernel<<<(wavesP + 3) / 4, 128, 0, stream>>>(targ,  packTarg,  B, V);
    pack_y_kernel<<<(wavesV + 3) / 4, 128, 0, stream>>>(dpred, packDPred, B, Nv);
    pack_y_kernel<<<(wavesV + 3) / 4, 128, 0, stream>>>(dtarg, packDTarg, B, Nv);
  }

  // chamfer (both directions) on positions and velocities
  {
    int groupsP = (tilesP + MT - 1) / MT;
    int wavesP  = B * groupsP;
    int blocksP = (wavesP + 3) / 4;
    chamfer_rowmin_kernel<<<blocksP, 128, 0, stream>>>(pred, packTarg, B, V, V, &acc[0]);
    chamfer_rowmin_kernel<<<blocksP, 128, 0, stream>>>(targ, packPred, B, V, V, &acc[0]);

    int groupsV = (tilesV + MT - 1) / MT;
    int wavesV  = B * groupsV;
    int blocksV = (wavesV + 3) / 4;
    chamfer_rowmin_kernel<<<blocksV, 128, 0, stream>>>(dpred, packDTarg, B, Nv, Nv, &acc[1]);
    chamfer_rowmin_kernel<<<blocksV, 128, 0, stream>>>(dtarg, packDPred, B, Nv, Nv, &acc[1]);
  }

  edge_kernel<<<(B * E + 255) / 256, 256, 0, stream>>>(pred, edges, B, V, E, &acc[2]);
  normal_kernel<<<(B * P + 255) / 256, 256, 0, stream>>>(pred, pairs, B, V, P, &acc[3]);
  lap_face_kernel<<<(B * F + 255) / 256, 256, 0, stream>>>(pred, faces, B, V, F, Lx, rs);
  lap_final_kernel<<<(B * V + 255) / 256, 256, 0, stream>>>(pred, Lx, rs, B, V, &acc[4]);

  final_kernel<<<1, 1, 0, stream>>>(acc, (float*)d_out, B, V, E, P);
}
